// SA_Layer_18116172054846
// MI455X (gfx1250) — matrix operand order repaired and checked by reading, not compiled, not run on hardware
//
#include <hip/hip_runtime.h>
#include <hip/hip_bf16.h>
#include <math.h>

#define BATCH 16
#define CH    256
#define NPTS  2048

typedef __bf16 bf16_t;
typedef bf16_t v16bf __attribute__((ext_vector_type(16)));
typedef bf16_t v8bf  __attribute__((ext_vector_type(8)));
typedef float  v8f   __attribute__((ext_vector_type(8)));

// ---------------------------------------------------------------------------
// Fragment loaders (wave32 WMMA 16x16x32 bf16 operands; this code's k order is contiguous, k = 16*(lane/16) + s, for A and B)
//
// A matrix 16x32 (MxK), lane l holds row M = l&15:
//   lanes 0-15 : VGPR0..3 = K 0..7,  VGPR4..7 = K 8..15   (kbase = 0)
//   lanes 16-31: VGPR0..3 = K 16..23, VGPR4..7 = K 24..31   (kbase = 8)
// -> one contiguous 16-element (32B) chunk at +2*kbase (two 16B loads).
//
// B matrix 32x16 (KxN), lane l holds column N = l&15:
//   lanes 0-15 : K 0..15 ; lanes 16-31: K 16..31  (koff = 0 / 16)
// -> one contiguous 16-element (32B) chunk at +koff.
// ---------------------------------------------------------------------------
__device__ __forceinline__ v16bf load_frag_a(const bf16_t* rowptr, int kbase) {
    v8bf lo = *(const v8bf*)(rowptr + 2 * kbase);
    v8bf hi = *(const v8bf*)(rowptr + 2 * kbase + 8);
    v16bf r;
#pragma unroll
    for (int i = 0; i < 8; ++i) { r[i] = lo[i]; r[i + 8] = hi[i]; }
    return r;
}

__device__ __forceinline__ v16bf load_frag_b(const bf16_t* colptr) {
    v8bf lo = *(const v8bf*)(colptr);
    v8bf hi = *(const v8bf*)(colptr + 8);
    v16bf r;
#pragma unroll
    for (int i = 0; i < 8; ++i) { r[i] = lo[i]; r[i + 8] = hi[i]; }
    return r;
}

#define WMMA_BF16(a, b, c) \
    __builtin_amdgcn_wmma_f32_16x16x32_bf16(false, (a), false, (b), (short)0, (c), false, false)

// ---------------------------------------------------------------------------
// K0a: convert weights f32 -> bf16
// ---------------------------------------------------------------------------
__global__ __launch_bounds__(256) void cvt_w_kernel(const float* __restrict__ wqk,
                                                    const float* __restrict__ wv,
                                                    bf16_t* __restrict__ wqkb,
                                                    bf16_t* __restrict__ wvb) {
    int i = blockIdx.x * 256 + threadIdx.x;           // 0 .. CH*CH-1
    wqkb[i] = (bf16_t)wqk[i];
    wvb[i]  = (bf16_t)wv[i];
}

// ---------------------------------------------------------------------------
// K0b: x [B,C,N] f32  ->  xT [B,N,C] bf16  (so GEMM B-operands are contiguous)
// ---------------------------------------------------------------------------
__global__ __launch_bounds__(256) void cvt_x_kernel(const float* __restrict__ x,
                                                    bf16_t* __restrict__ xT) {
    long idx = (long)blockIdx.x * 256 + threadIdx.x;  // < B*C*N  (coalesced read)
    int  n   = (int)(idx % NPTS);
    long t   = idx / NPTS;
    int  c   = (int)(t % CH);
    int  b   = (int)(t / CH);
    xT[((long)b * NPTS + n) * CH + c] = (bf16_t)x[idx];
}

// ---------------------------------------------------------------------------
// K1: projections.  blockIdx.z = b*2 + which
//   which==0: yT[b,n,o] = sum_c Wqk[o,c] * x[b,c,n]          (stored [B,N,C] bf16)
//   which==1: v [b,c,n] = sum_c' Wv[c,c'] * x[b,c',n] + bv   (stored [B,C,N] bf16)
// One 16x16 D tile per wave, K = CH = 256 -> 8 WMMAs.
// ---------------------------------------------------------------------------
__global__ __launch_bounds__(256) void proj_kernel(const bf16_t* __restrict__ wqkb,
                                                   const bf16_t* __restrict__ wvb,
                                                   const bf16_t* __restrict__ xT,
                                                   const float*  __restrict__ bv,
                                                   bf16_t* __restrict__ yT,
                                                   bf16_t* __restrict__ v) {
    const int wave  = threadIdx.x >> 5;
    const int lane  = threadIdx.x & 31;
    const int z     = blockIdx.z;
    const int b     = z >> 1;
    const bool is_v = (z & 1) != 0;
    const int ntile = blockIdx.x * 8 + wave;          // 0..127 (N tiles)
    const int mtile = blockIdx.y;                     // 0..15  (output-channel tiles)

    const int l15   = lane & 15;
    const int kbase = (lane < 16) ? 0 : 8;
    const int koff  = (lane < 16) ? 0 : 16;
    const int mbase = (lane < 16) ? 0 : 8;

    const bf16_t* W    = is_v ? wvb : wqkb;
    const int  orow    = mtile * 16 + l15;
    const long ncol    = (long)ntile * 16 + l15;
    const bf16_t* arow = W + (long)orow * CH;
    const bf16_t* brow = xT + ((long)b * NPTS + ncol) * CH;

    v8f acc = {};
#pragma unroll
    for (int kk = 0; kk < CH; kk += 32) {
        v16bf a  = load_frag_a(arow + kk, kbase);
        v16bf bm = load_frag_b(brow + kk + koff);
        acc = WMMA_BF16(a, bm, acc);
    }

    if (!is_v) {
        // transposed store: yT[(b*N+n)*C + o], lane's 8 rows are contiguous in o
        bf16_t* dst = yT + ((long)b * NPTS + ncol) * CH + mtile * 16 + mbase;
        v8bf pk;
#pragma unroll
        for (int r = 0; r < 8; ++r) pk[r] = (bf16_t)acc[r];
        *(v8bf*)dst = pk;
    } else {
        // natural [C,N] store with bias epilogue
#pragma unroll
        for (int r = 0; r < 8; ++r) {
            int c = mtile * 16 + mbase + r;
            v[((long)b * CH + c) * NPTS + ncol] = (bf16_t)(acc[r] + bv[c]);
        }
    }
}

// ---------------------------------------------------------------------------
// K2: energy[n,m] = sum_o yT[n,o]*yT[m,o]  (Gram matrix -> SYMMETRIC).
// Per batch b. Each wave computes a 32x32 super-tile (2x2 register blocking,
// 4 accumulators, 2 loads per WMMA) of the UPPER triangle only and mirror-
// stores the transpose block. Mirror store is per-lane contiguous (32B).
// Super-tiles per dim: 64; upper-tri pairs incl diag: 64*65/2 = 2080 waves.
// ---------------------------------------------------------------------------
__global__ __launch_bounds__(256) void energy_kernel(const bf16_t* __restrict__ yT,
                                                     float* __restrict__ energy,
                                                     int b) {
    const int wave = threadIdx.x >> 5;
    const int lane = threadIdx.x & 31;
    int p = blockIdx.x * 8 + wave;                    // 0..2079 (pair index)

    // decode upper-triangular pair p -> (i, j), i <= j (row-major enumeration)
    int i = 0;
    while (p >= (64 - i)) { p -= (64 - i); ++i; }
    const int j = i + p;

    const int n0 = i * 32;                            // row block (n)
    const int m0 = j * 32;                            // col block (m)

    const int l15   = lane & 15;
    const int kbase = (lane < 16) ? 0 : 8;
    const int koff  = (lane < 16) ? 0 : 16;
    const int mbase = (lane < 16) ? 0 : 8;

    const bf16_t* arow0 = yT + ((long)b * NPTS + n0 + l15) * CH;
    const bf16_t* arow1 = arow0 + (long)16 * CH;
    const bf16_t* brow0 = yT + ((long)b * NPTS + m0 + l15) * CH;
    const bf16_t* brow1 = brow0 + (long)16 * CH;

    v8f acc00 = {}, acc01 = {}, acc10 = {}, acc11 = {};
#pragma unroll
    for (int kk = 0; kk < CH; kk += 32) {
        v16bf a0 = load_frag_a(arow0 + kk, kbase);
        v16bf a1 = load_frag_a(arow1 + kk, kbase);
        v16bf b0 = load_frag_b(brow0 + kk + koff);
        v16bf b1 = load_frag_b(brow1 + kk + koff);
        acc00 = WMMA_BF16(a0, b0, acc00);
        acc01 = WMMA_BF16(a0, b1, acc01);
        acc10 = WMMA_BF16(a1, b0, acc10);
        acc11 = WMMA_BF16(a1, b1, acc11);
    }

    // direct stores: energy[n, m] (scattered f32 per lane)
#pragma unroll
    for (int tn = 0; tn < 2; ++tn) {
#pragma unroll
        for (int tm = 0; tm < 2; ++tm) {
            const v8f acc = (tn == 0) ? ((tm == 0) ? acc00 : acc01)
                                      : ((tm == 0) ? acc10 : acc11);
            const int m = m0 + tm * 16 + l15;
#pragma unroll
            for (int r = 0; r < 8; ++r) {
                int n = n0 + tn * 16 + mbase + r;
                energy[(long)n * NPTS + m] = acc[r];
            }
        }
    }

    // mirror stores: energy[m, n] = energy[n, m] (contiguous 8 f32 per lane)
    if (i != j) {
#pragma unroll
        for (int tn = 0; tn < 2; ++tn) {
#pragma unroll
            for (int tm = 0; tm < 2; ++tm) {
                const v8f acc = (tn == 0) ? ((tm == 0) ? acc00 : acc01)
                                          : ((tm == 0) ? acc10 : acc11);
                const int m = m0 + tm * 16 + l15;
                *(v8f*)(energy + (long)m * NPTS + n0 + tn * 16 + mbase) = acc;
            }
        }
    }
}

// ---------------------------------------------------------------------------
// K3: row softmax of energy; writes attT[m,n] = softmax(energy[n,:])[m] in bf16
// (transposed so the out-GEMM B operand and the column-sum pass are contiguous)
// ---------------------------------------------------------------------------
__global__ __launch_bounds__(256) void softmax_kernel(const float* __restrict__ energy,
                                                      bf16_t* __restrict__ attT) {
    __shared__ float red[256];
    const int n   = blockIdx.x;
    const int tid = threadIdx.x;
    const float* row = energy + (long)n * NPTS;

    float mx = -3.402823466e38f;
    for (int m = tid; m < NPTS; m += 256) mx = fmaxf(mx, row[m]);
    red[tid] = mx;
    __syncthreads();
    for (int off = 128; off > 0; off >>= 1) {
        if (tid < off) red[tid] = fmaxf(red[tid], red[tid + off]);
        __syncthreads();
    }
    const float gmax = red[0];
    __syncthreads();

    float sm = 0.0f;
    for (int m = tid; m < NPTS; m += 256) sm += __expf(row[m] - gmax);
    red[tid] = sm;
    __syncthreads();
    for (int off = 128; off > 0; off >>= 1) {
        if (tid < off) red[tid] += red[tid + off];
        __syncthreads();
    }
    const float inv = 1.0f / red[0];

    for (int m = tid; m < NPTS; m += 256)
        attT[(long)m * NPTS + n] = (bf16_t)(__expf(row[m] - gmax) * inv);
}

// ---------------------------------------------------------------------------
// K4: per-column scale s[m] = 1 / ((1e-9 + sum_n att[n,m]) * sqrt(C))
// attT row m is contiguous -> coalesced reduction.
// ---------------------------------------------------------------------------
__global__ __launch_bounds__(256) void scale_kernel(const bf16_t* __restrict__ attT,
                                                    float* __restrict__ s) {
    __shared__ float red[256];
    const int m   = blockIdx.x;
    const int tid = threadIdx.x;
    const bf16_t* row = attT + (long)m * NPTS;

    float sm = 0.0f;
    for (int n = tid; n < NPTS; n += 256) sm += (float)row[n];
    red[tid] = sm;
    __syncthreads();
    for (int off = 128; off > 0; off >>= 1) {
        if (tid < off) red[tid] += red[tid + off];
        __syncthreads();
    }
    if (tid == 0) s[m] = 1.0f / ((1e-9f + red[0]) * 16.0f);   // sqrt(256) = 16
}

// ---------------------------------------------------------------------------
// K5: out[b,c,m] = (sum_n v[b,c,n] * att[n,m]) * s[m].  K = NPTS = 2048.
// 2x2 register blocking: wave computes a 32(c) x 32(m) block -> 4 accumulators,
// 2 loads per WMMA. A = rows c of v (K=n contiguous); B col m = attT row m
// (contiguous). Column renorm + 1/sqrt(C) folded into epilogue scale s[m].
// ---------------------------------------------------------------------------
__global__ __launch_bounds__(256) void out_kernel(const bf16_t* __restrict__ v,
                                                  const bf16_t* __restrict__ attT,
                                                  const float*  __restrict__ s,
                                                  float* __restrict__ out,
                                                  int b) {
    const int wave = threadIdx.x >> 5;
    const int lane = threadIdx.x & 31;
    const int msup = blockIdx.x * 8 + wave;           // 0..63  (32-wide m blocks)
    const int csup = blockIdx.y;                      // 0..7   (32-wide c blocks)
    const int m0   = msup * 32;
    const int c0   = csup * 32;

    const int l15   = lane & 15;
    const int kbase = (lane < 16) ? 0 : 8;
    const int koff  = (lane < 16) ? 0 : 16;
    const int mbase = (lane < 16) ? 0 : 8;

    const bf16_t* arow0 = v + ((long)b * CH + c0 + l15) * NPTS;
    const bf16_t* arow1 = arow0 + (long)16 * NPTS;
    const bf16_t* brow0 = attT + (long)(m0 + l15) * NPTS;
    const bf16_t* brow1 = brow0 + (long)16 * NPTS;

    v8f acc00 = {}, acc01 = {}, acc10 = {}, acc11 = {};
    for (int kk = 0; kk < NPTS; kk += 32) {
        v16bf a0 = load_frag_a(arow0 + kk, kbase);
        v16bf a1 = load_frag_a(arow1 + kk, kbase);
        v16bf b0 = load_frag_b(brow0 + kk + koff);
        v16bf b1 = load_frag_b(brow1 + kk + koff);
        acc00 = WMMA_BF16(a0, b0, acc00);
        acc01 = WMMA_BF16(a0, b1, acc01);
        acc10 = WMMA_BF16(a1, b0, acc10);
        acc11 = WMMA_BF16(a1, b1, acc11);
    }

    const float sc0 = s[m0 + l15];
    const float sc1 = s[m0 + 16 + l15];
#pragma unroll
    for (int tn = 0; tn < 2; ++tn) {
#pragma unroll
        for (int tm = 0; tm < 2; ++tm) {
            const v8f   acc = (tn == 0) ? ((tm == 0) ? acc00 : acc01)
                                        : ((tm == 0) ? acc10 : acc11);
            const float sc  = (tm == 0) ? sc0 : sc1;
            const int   m   = m0 + tm * 16 + l15;
#pragma unroll
            for (int r = 0; r < 8; ++r) {
                int c = c0 + tn * 16 + mbase + r;
                out[((long)b * CH + c) * NPTS + m] = acc[r] * sc;
            }
        }
    }
}

// ---------------------------------------------------------------------------
// Workspace layout (bytes):
//   xT    : B*N*C*2  = 16,777,216
//   wqkb  : C*C*2    =    131,072
//   wvb   : C*C*2    =    131,072
//   yT    : B*N*C*2  = 16,777,216
//   v     : B*C*N*2  = 16,777,216
//   energy: N*N*4    = 16,777,216   (per-batch, reused)
//   attT  : N*N*2    =  8,388,608   (per-batch, reused)
//   s     : N*4      =      8,192   (per-batch, reused)
//   total ~ 75.8 MB
// ---------------------------------------------------------------------------
extern "C" void kernel_launch(void* const* d_in, const int* in_sizes, int n_in,
                              void* d_out, int out_size, void* d_ws, size_t ws_size,
                              hipStream_t stream) {
    const float* x   = (const float*)d_in[0];
    const float* wqk = (const float*)d_in[1];
    const float* wv  = (const float*)d_in[2];
    const float* bv  = (const float*)d_in[3];
    float* out = (float*)d_out;

    char* ws = (char*)d_ws;
    const size_t SZ_XT  = (size_t)BATCH * NPTS * CH * sizeof(bf16_t);
    const size_t SZ_W   = (size_t)CH * CH * sizeof(bf16_t);
    const size_t SZ_YT  = SZ_XT;
    const size_t SZ_V   = SZ_XT;
    const size_t SZ_E   = (size_t)NPTS * NPTS * sizeof(float);
    const size_t SZ_AT  = (size_t)NPTS * NPTS * sizeof(bf16_t);

    bf16_t* xT     = (bf16_t*)(ws);
    bf16_t* wqkb   = (bf16_t*)(ws + SZ_XT);
    bf16_t* wvb    = (bf16_t*)(ws + SZ_XT + SZ_W);
    bf16_t* yT     = (bf16_t*)(ws + SZ_XT + 2 * SZ_W);
    bf16_t* v      = (bf16_t*)(ws + SZ_XT + 2 * SZ_W + SZ_YT);
    float*  energy = (float*) (ws + SZ_XT + 2 * SZ_W + SZ_YT + SZ_V);
    bf16_t* attT   = (bf16_t*)(ws + SZ_XT + 2 * SZ_W + SZ_YT + SZ_V + SZ_E);
    float*  s      = (float*) (ws + SZ_XT + 2 * SZ_W + SZ_YT + SZ_V + SZ_E + SZ_AT);

    // K0: conversions / transposes
    cvt_w_kernel<<<(CH * CH) / 256, 256, 0, stream>>>(wqk, wv, wqkb, wvb);
    cvt_x_kernel<<<(BATCH * CH * NPTS) / 256, 256, 0, stream>>>(x, xT);

    // K1: both projections for all batches (z = b*2 + which)
    proj_kernel<<<dim3(16, 16, BATCH * 2), 256, 0, stream>>>(wqkb, wvb, xT, bv, yT, v);

    // Per-batch attention pipeline (reuses energy/attT/s buffers)
    for (int b = 0; b < BATCH; ++b) {
        energy_kernel<<<2080 / 8, 256, 0, stream>>>(yT, energy, b);
        softmax_kernel<<<NPTS, 256, 0, stream>>>(energy, attT);
        scale_kernel<<<NPTS, 256, 0, stream>>>(attT, s);
        out_kernel<<<dim3(8, 8), 256, 0, stream>>>(v, attT, s, out, b);
    }
}
